// LiDAREncoder_60670708023396
// MI455X (gfx1250) — matrix operand order repaired and checked by reading, not compiled, not run on hardware
//
#include <hip/hip_runtime.h>
#include <hip/hip_bf16.h>
#include <hip/hip_fp16.h>

// ---------------------------------------------------------------------------
// PointPillars-style LiDAR encoder for gfx1250 (MI455X).
// All GEMMs (pillar MLP + 3x3 convs as implicit GEMM) run on
// v_wmma_f32_16x16x32_f16 (f16 in, f32 accumulate).  Conv input tiles are
// staged into LDS by the Tensor Data Mover (tensor_load_to_lds +
// s_wait_tensorcnt); next-tap weights are prefetched with global_prefetch.
// ---------------------------------------------------------------------------

typedef __attribute__((ext_vector_type(16))) _Float16 v16h;
typedef __attribute__((ext_vector_type(8)))  float    v8f;
typedef __attribute__((ext_vector_type(4)))  unsigned u32x4;
typedef __attribute__((ext_vector_type(8)))  int      i32x8;
typedef __attribute__((ext_vector_type(4)))  int      i32x4;

#define BATCH   2
#define NPTS    60000
#define GNX     256
#define GNY     256
#define MAXPP   32
#define MAXP    12000
#define SENTID  (GNX*GNY)          // 65536
#define ROWS    (MAXP*MAXPP)       // 384000 rows per batch in the MLP
#define TILES16 (ROWS/16)          // 24000
#define PADW    258                // 256 + 1-pixel zero halo each side
#define PCR0f   (-32.0f)
#define PS0f    (0.25f)

// ------------------------------- utility -----------------------------------

__global__ void k_fill_u32(unsigned* __restrict__ p, unsigned v, long n) {
  long i = (long)blockIdx.x * blockDim.x + threadIdx.x;
  long stride = (long)gridDim.x * blockDim.x;
  for (; i < n; i += stride) p[i] = v;
}

// ----------------------------- pillarization -------------------------------

__global__ void k1_assign(const float* __restrict__ pts,
                          const unsigned char* __restrict__ msk,
                          int* __restrict__ pid, unsigned* __restrict__ occ) {
  int i = blockIdx.x * blockDim.x + threadIdx.x;
  if (i >= BATCH * NPTS) return;
  int b = i / NPTS;
  const float* pt = pts + (size_t)i * 5;
  int id = SENTID;
  if (msk[i]) {
    int ix = (int)((pt[0] - PCR0f) / PS0f);
    int iy = (int)((pt[1] - PCR0f) / PS0f);
    ix = min(max(ix, 0), GNX - 1);
    iy = min(max(iy, 0), GNY - 1);
    id = iy * GNX + ix;
    occ[b * SENTID + id] = 1u;   // idempotent
  }
  pid[i] = id;
}

// one workgroup per batch: exclusive scan over 65536 cells -> pillar slots
__global__ __launch_bounds__(1024) void k2_scan(const unsigned* __restrict__ occ,
                                                int* __restrict__ cell2slot,
                                                int* __restrict__ slot2cell) {
  int b = blockIdx.x, tid = threadIdx.x;
  __shared__ unsigned sc[1024];
  __shared__ unsigned sbase;
  if (tid == 0) sbase = 0;
  __syncthreads();
  for (int chunk = 0; chunk < SENTID / 1024; ++chunk) {
    int cell = chunk * 1024 + tid;
    unsigned v = occ[b * SENTID + cell];
    sc[tid] = v;
    __syncthreads();
    for (int off = 1; off < 1024; off <<= 1) {
      unsigned t = (tid >= off) ? sc[tid - off] : 0u;
      __syncthreads();
      sc[tid] += t;
      __syncthreads();
    }
    unsigned incl = sc[tid];
    unsigned rank = sbase + incl - v;
    int slot = (v && rank < MAXP) ? (int)rank : -1;
    cell2slot[b * SENTID + cell] = slot;
    if (slot >= 0) slot2cell[b * MAXP + slot] = cell;
    __syncthreads();
    if (tid == 1023) sbase += incl;
    __syncthreads();
  }
}

__global__ void k3_rank(const float* __restrict__ pts, const int* __restrict__ pid,
                        const int* __restrict__ cell2slot,
                        int* __restrict__ arrive, int* __restrict__ kcnt,
                        float* __restrict__ sums,
                        int* __restrict__ pslot, int* __restrict__ prank) {
  int i = blockIdx.x * blockDim.x + threadIdx.x;
  if (i >= BATCH * NPTS) return;
  int b = i / NPTS;
  int id = pid[i];
  int slot = -1, r = -1;
  if (id < SENTID) {
    int s = cell2slot[b * SENTID + id];
    if (s >= 0) {
      r = atomicAdd(&arrive[b * MAXP + s], 1);
      if (r < MAXPP) {
        slot = s;
        const float* pt = pts + (size_t)i * 5;
        atomicAdd(&kcnt[b * MAXP + s], 1);
        atomicAdd(&sums[(b * MAXP + s) * 3 + 0], pt[0]);
        atomicAdd(&sums[(b * MAXP + s) * 3 + 1], pt[1]);
        atomicAdd(&sums[(b * MAXP + s) * 3 + 2], pt[2]);
      }
    }
  }
  pslot[i] = slot;
  prank[i] = r;
}

__global__ void k4_features(const float* __restrict__ pts, const int* __restrict__ pid,
                            const int* __restrict__ pslot, const int* __restrict__ prank,
                            const int* __restrict__ kcnt, const float* __restrict__ sums,
                            float* __restrict__ pf, unsigned* __restrict__ pm) {
  int i = blockIdx.x * blockDim.x + threadIdx.x;
  if (i >= BATCH * NPTS) return;
  int s = pslot[i];
  if (s < 0) return;
  int b = i / NPTS;
  int r = prank[i];
  int idx = b * MAXP + s;
  const float* pt = pts + (size_t)i * 5;
  float cnt = (float)max(kcnt[idx], 1);
  float cx = sums[idx * 3 + 0] / cnt;
  float cy = sums[idx * 3 + 1] / cnt;
  float cz = sums[idx * 3 + 2] / cnt;
  int id = pid[i];
  float gx = PCR0f + ((float)(id & (GNX - 1)) + 0.5f) * PS0f;
  float gy = PCR0f + ((float)(id >> 8) + 0.5f) * PS0f;
  float* dst = pf + ((size_t)idx * MAXPP + r) * 10;
  dst[0] = pt[0]; dst[1] = pt[1]; dst[2] = pt[2]; dst[3] = pt[3]; dst[4] = pt[4];
  dst[5] = pt[0] - cx; dst[6] = pt[1] - cy; dst[7] = pt[2] - cz;
  dst[8] = gx; dst[9] = gy;
  atomicOr(&pm[idx], 1u << r);
}

// ------------------------- weight conversion to f16 -------------------------
// layouts: w1h[n*32+k] (k>=10 zero-padded), w2h[n*64+k], cwh[l][t][n*64+k]

__global__ void kw_prep(const float* __restrict__ w1, const float* __restrict__ w2,
                        const float* __restrict__ cw1, const float* __restrict__ cw2,
                        const float* __restrict__ cw3,
                        _Float16* __restrict__ w1h, _Float16* __restrict__ w2h,
                        _Float16* __restrict__ cwh) {
  int i = blockIdx.x * blockDim.x + threadIdx.x;
  if (i < 2048) {
    int n = i >> 5, k = i & 31;
    w1h[i] = (k < 10) ? (_Float16)w1[k * 64 + n] : (_Float16)0.0f;
  } else if (i < 2048 + 4096) {
    int j = i - 2048;
    int n = j >> 6, k = j & 63;
    w2h[j] = (_Float16)w2[k * 64 + n];
  } else if (i < 2048 + 4096 + 3 * 9 * 4096) {
    int j = i - 6144;
    int l = j / 36864;
    int jj = j - l * 36864;
    int t = jj >> 12;
    int rest = jj & 4095;
    int n = rest >> 6, k = rest & 63;
    const float* cw = (l == 0) ? cw1 : ((l == 1) ? cw2 : cw3);
    int ky = t / 3, kx = t - ky * 3;
    cwh[j] = (_Float16)cw[((n * 64 + k) * 3 + ky) * 3 + kx];   // OIHW
  }
}

// ----------------------------- WMMA helpers ---------------------------------
// A fragment (16x32 f16): lane L holds row M=L&15; element pair (2j,2j+1) has
// k = 2j + 16*(L>>4) (+1).  B fragment: with weights stored [n][K]
// each lane reads 16 contiguous halves at n*K + 32*kc + 16*(L>>4).

__device__ __forceinline__ v16h afrag_from_f32row(const float* rowp, int laneHiK, int kbase) {
  v16h a;
#pragma unroll
  for (int j = 0; j < 8; ++j) {
    int klo = 2 * j + 2 * laneHiK + kbase;
    float2 v = *(const float2*)(rowp + klo);
    a[2 * j] = (_Float16)v.x;
    a[2 * j + 1] = (_Float16)v.y;
  }
  return a;
}

// --------------------------- MLP layer 1 (K=10->32) --------------------------

__global__ __launch_bounds__(128) void k5_gemm1(const float* __restrict__ pf,
                                                const _Float16* __restrict__ w1h,
                                                const float* __restrict__ b1,
                                                float* __restrict__ y1,
                                                float* __restrict__ bn1acc) {
  const int b = blockIdx.y;
  const int warp = threadIdx.x >> 5, lane = threadIdx.x & 31;
  const int laneLo = lane & 15, laneHi8 = (lane >> 4) * 8;
  const int tile = blockIdx.x * 4 + warp;
  __shared__ float ssum[64], ssq[64];
  if (threadIdx.x < 64) { ssum[threadIdx.x] = 0.f; ssq[threadIdx.x] = 0.f; }
  __syncthreads();

  const float* pfb = pf + (size_t)b * MAXP * MAXPP * 10;
  float* y1b = y1 + (size_t)b * ROWS * 64;
  const float* prow = pfb + (size_t)(tile * 16 + laneLo) * 10;

  v16h a;
#pragma unroll
  for (int j = 0; j < 8; ++j) {
    int klo = 2 * j + 2 * laneHi8;
    float v0 = (klo < 10) ? prow[klo] : 0.f;
    float v1 = (klo + 1 < 10) ? prow[klo + 1] : 0.f;
    a[2 * j] = (_Float16)v0;
    a[2 * j + 1] = (_Float16)v1;
  }
#pragma unroll
  for (int nt = 0; nt < 4; ++nt) {
    int ch = nt * 16 + laneLo;
    v16h bf = *(const v16h*)(w1h + ch * 32 + ((lane >> 4) << 4));
    v8f c = {};
    c = __builtin_amdgcn_wmma_f32_16x16x32_f16(false, a, false, bf, (short)0, c, false, false);
    float bias = b1[ch];
    float lsum = 0.f, lsq = 0.f;
#pragma unroll
    for (int r = 0; r < 8; ++r) {
      float v = c[r] + bias;
      y1b[(size_t)(tile * 16 + r + laneHi8) * 64 + ch] = v;
      lsum += v; lsq += v * v;
    }
    atomicAdd(&ssum[ch], lsum);
    atomicAdd(&ssq[ch], lsq);
  }
  __syncthreads();
  if (threadIdx.x < 64) {
    atomicAdd(&bn1acc[(b * 64 + threadIdx.x) * 2 + 0], ssum[threadIdx.x]);
    atomicAdd(&bn1acc[(b * 64 + threadIdx.x) * 2 + 1], ssq[threadIdx.x]);
  }
}

// BN finalize: scale = g*rsqrt(var+eps), shift = be - mean*scale
__global__ void k_bnfin(const float* __restrict__ acc, const float* __restrict__ g,
                        const float* __restrict__ be, float* __restrict__ scsh,
                        float invN, int n) {
  int i = blockIdx.x * blockDim.x + threadIdx.x;
  if (i >= n) return;
  int ch = i & 63;
  float s = acc[i * 2], q = acc[i * 2 + 1];
  float m = s * invN;
  float v = q * invN - m * m;
  float sc = g[ch] * rsqrtf(v + 1e-5f);
  scsh[i * 2] = sc;
  scsh[i * 2 + 1] = be[ch] - m * sc;
}

// --------------------------- MLP layer 2 (K=64) ------------------------------

__global__ __launch_bounds__(128) void k7_gemm2(const float* __restrict__ y1,
                                                const _Float16* __restrict__ w2h,
                                                const float* __restrict__ b2,
                                                const float* __restrict__ bn1p,
                                                float* __restrict__ y2,
                                                float* __restrict__ bn2acc) {
  const int b = blockIdx.y;
  const int warp = threadIdx.x >> 5, lane = threadIdx.x & 31;
  const int laneLo = lane & 15, laneHi8 = (lane >> 4) * 8;
  const int tile = blockIdx.x * 4 + warp;
  __shared__ float s_sc[64], s_sh[64], ssum[64], ssq[64];
  if (threadIdx.x < 64) {
    s_sc[threadIdx.x] = bn1p[(b * 64 + threadIdx.x) * 2 + 0];
    s_sh[threadIdx.x] = bn1p[(b * 64 + threadIdx.x) * 2 + 1];
    ssum[threadIdx.x] = 0.f; ssq[threadIdx.x] = 0.f;
  }
  __syncthreads();

  const float* y1row = y1 + (size_t)b * ROWS * 64 + (size_t)(tile * 16 + laneLo) * 64;
  float* y2b = y2 + (size_t)b * ROWS * 64;
  v8f c[4] = {};
#pragma unroll
  for (int kc = 0; kc < 2; ++kc) {
    v16h a;
#pragma unroll
    for (int j = 0; j < 8; ++j) {
      int klo = 2 * j + 2 * laneHi8 + kc * 32;
      float2 v = *(const float2*)(y1row + klo);
      float h0 = fmaxf(fmaf(v.x, s_sc[klo], s_sh[klo]), 0.f);
      float h1 = fmaxf(fmaf(v.y, s_sc[klo + 1], s_sh[klo + 1]), 0.f);
      a[2 * j] = (_Float16)h0;
      a[2 * j + 1] = (_Float16)h1;
    }
#pragma unroll
    for (int nt = 0; nt < 4; ++nt) {
      v16h bf = *(const v16h*)(w2h + (nt * 16 + laneLo) * 64 + kc * 32 + ((lane >> 4) << 4));
      c[nt] = __builtin_amdgcn_wmma_f32_16x16x32_f16(false, a, false, bf, (short)0, c[nt], false, false);
    }
  }
#pragma unroll
  for (int nt = 0; nt < 4; ++nt) {
    int ch = nt * 16 + laneLo;
    float bias = b2[ch];
    float lsum = 0.f, lsq = 0.f;
#pragma unroll
    for (int r = 0; r < 8; ++r) {
      float v = c[nt][r] + bias;
      y2b[(size_t)(tile * 16 + r + laneHi8) * 64 + ch] = v;
      lsum += v; lsq += v * v;
    }
    atomicAdd(&ssum[ch], lsum);
    atomicAdd(&ssq[ch], lsq);
  }
  __syncthreads();
  if (threadIdx.x < 64) {
    atomicAdd(&bn2acc[(b * 64 + threadIdx.x) * 2 + 0], ssum[threadIdx.x]);
    atomicAdd(&bn2acc[(b * 64 + threadIdx.x) * 2 + 1], ssq[threadIdx.x]);
  }
}

// ------------------- masked max over 32 points + BEV scatter -----------------

__global__ __launch_bounds__(64) void k9_maxscatter(const float* __restrict__ y2,
                                                    const unsigned* __restrict__ pm,
                                                    const int* __restrict__ slot2cell,
                                                    const float* __restrict__ bn2p,
                                                    float* __restrict__ actA) {
  int b = blockIdx.y, slot = blockIdx.x, ch = threadIdx.x;
  int cell = slot2cell[b * MAXP + slot];
  if (cell < 0) return;
  unsigned mask = pm[b * MAXP + slot];
  float sc = bn2p[(b * 64 + ch) * 2 + 0];
  float sh = bn2p[(b * 64 + ch) * 2 + 1];
  const float* yb = y2 + ((size_t)b * ROWS + (size_t)slot * MAXPP) * 64;
  float m = -1e9f;
  for (int r = 0; r < MAXPP; ++r) {
    if (mask & (1u << r)) {
      float v = fmaxf(fmaf(yb[r * 64 + ch], sc, sh), 0.f);
      m = fmaxf(m, v);
    }
  }
  int x = cell & (GNX - 1);
  int y = cell >> 8;
  actA[(((size_t)b * PADW + (y + 1)) * PADW + (x + 1)) * 64 + ch] = m;
}

// --------------------------- 3x3 conv via WMMA -------------------------------
// workgroup = 8 waves, computes 8(H)x16(W)x64(C) output tile; input window
// 10x18x64 f32 staged in LDS by the Tensor Data Mover (6-arg builtin form).

__global__ __launch_bounds__(256) void k10_conv(const float* __restrict__ actIn,
                                                const _Float16* __restrict__ wtap,
                                                const float* __restrict__ cb,
                                                float* __restrict__ rawOut,
                                                float* __restrict__ bnacc) {
  const int x0 = blockIdx.x * 16, y0 = blockIdx.y * 8, b = blockIdx.z;
  const int tid = threadIdx.x, warp = tid >> 5, lane = tid & 31;
  const int laneLo = lane & 15, laneHi8 = (lane >> 4) * 8;

  __shared__ __align__(16) float s_in[10 * 18 * 64];   // 46080 B
  __shared__ __align__(32) _Float16 s_w[64 * 64];      //  8192 B
  __shared__ float ssum[64], ssq[64];
  if (tid < 64) { ssum[tid] = 0.f; ssq[tid] = 0.f; }

  // padded window start: rows y0..y0+9, cols x0..x0+17 (all in-bounds)
  const size_t winElem = (((size_t)b * PADW + y0) * PADW + x0) * 64;

  // Tensor Data Mover: 2D tile, 10 rows of 18*64 f32, line stride 258*64 f32.
  if (warp == 0) {
    unsigned long long ga = (unsigned long long)(const void*)(actIn + winElem);
    unsigned ldsa = (unsigned)(unsigned long long)(const void*)&s_in[0];
    const unsigned td0 = PADW * 64;        // tensor line length (elements)
    const unsigned td1 = 1u << 16;         // large; window fully in-bounds
    const unsigned tile0 = 18 * 64;        // window line (elements)
    const unsigned tile1 = 10;             // rows
    const unsigned long long str0 = PADW * 64;
    u32x4 g0;
    g0[0] = 1u;                                             // count=1, user D#
    g0[1] = ldsa;                                           // LDS byte address
    g0[2] = (unsigned)(ga & 0xffffffffu);                   // global addr lo
    g0[3] = (unsigned)((ga >> 32) & 0x01ffffffu) | (2u << 30); // addr hi | type=2
    i32x8 g1;
    g1[0] = (int)(2u << 16);                                // data_size=4B
    g1[1] = (int)((td0 & 0xffffu) << 16);                   // tensor_dim0[15:0]
    g1[2] = (int)((td0 >> 16) | ((td1 & 0xffffu) << 16));   // td0 hi | td1 lo
    g1[3] = (int)((td1 >> 16) | (tile0 << 16));             // td1 hi | tile_dim0
    g1[4] = (int)tile1;                                     // tile_dim1 (tile_dim2=0)
    g1[5] = (int)(str0 & 0xffffffffu);                      // dim0 stride lo
    g1[6] = (int)((str0 >> 32) & 0xffffu);                  // dim0 stride hi
    g1[7] = 0;
    i32x4 g2 = {0, 0, 0, 0}, g3 = {0, 0, 0, 0};
    i32x8 g4 = {0, 0, 0, 0, 0, 0, 0, 0};
    __builtin_amdgcn_tensor_load_to_lds(g0, g1, g2, g3, g4, 0);
    __builtin_amdgcn_s_wait_tensorcnt(0);
  }
  __syncthreads();

  v8f c[4] = {};
  const int ytile = warp;   // output row within the tile
  for (int t = 0; t < 9; ++t) {
    __syncthreads();        // previous tap's compute done before restage
    {
      const unsigned* wsrc = (const unsigned*)(wtap + t * 4096);
      unsigned* wdst = (unsigned*)s_w;
      for (int i = tid; i < 2048; i += 256) wdst[i] = wsrc[i];
    }
    __syncthreads();
    if (t < 8)  // pull next tap's weight slice toward L2/L0 while we compute
      __builtin_prefetch(wtap + (t + 1) * 4096 + (tid & 31) * 128, 0, 3);
    const int ky = t / 3, kx = t - ky * 3;
    const float* rowp = &s_in[((ytile + ky) * 18 + (laneLo + kx)) * 64];
#pragma unroll
    for (int kc = 0; kc < 2; ++kc) {
      v16h a = afrag_from_f32row(rowp, laneHi8, kc * 32);
#pragma unroll
      for (int nt = 0; nt < 4; ++nt) {
        v16h bf = *(const v16h*)(s_w + (nt * 16 + laneLo) * 64 + kc * 32 + ((lane >> 4) << 4));
        c[nt] = __builtin_amdgcn_wmma_f32_16x16x32_f16(false, a, false, bf, (short)0, c[nt], false, false);
      }
    }
  }

  float* outr = rawOut + (((size_t)b * GNY + (y0 + ytile)) * GNX) * 64;
#pragma unroll
  for (int nt = 0; nt < 4; ++nt) {
    int ch = nt * 16 + laneLo;
    float bias = cb[ch];
    float lsum = 0.f, lsq = 0.f;
#pragma unroll
    for (int r = 0; r < 8; ++r) {
      int xo = x0 + r + laneHi8;
      float v = c[nt][r] + bias;
      outr[(size_t)xo * 64 + ch] = v;
      lsum += v; lsq += v * v;
    }
    atomicAdd(&ssum[ch], lsum);
    atomicAdd(&ssq[ch], lsq);
  }
  __syncthreads();
  if (tid < 64) {
    atomicAdd(&bnacc[tid * 2 + 0], ssum[tid]);
    atomicAdd(&bnacc[tid * 2 + 1], ssq[tid]);
  }
}

// ---------------------- BN apply (+ReLU) elementwise -------------------------

__global__ void k11_apply_pad(const float* __restrict__ raw, const float* __restrict__ scsh,
                              float* __restrict__ actOut) {
  long n = (long)BATCH * GNY * GNX * 64;
  long i = (long)blockIdx.x * blockDim.x + threadIdx.x;
  long stride = (long)gridDim.x * blockDim.x;
  for (; i < n; i += stride) {
    int ch = (int)(i & 63);
    int x = (int)((i >> 6) & 255);
    int y = (int)((i >> 14) & 255);
    int b = (int)(i >> 22);
    float v = fmaxf(fmaf(raw[i], scsh[ch * 2], scsh[ch * 2 + 1]), 0.f);
    actOut[(((size_t)b * PADW + (y + 1)) * PADW + (x + 1)) * 64 + ch] = v;
  }
}

__global__ void k11_apply_out(const float* __restrict__ raw, const float* __restrict__ scsh,
                              float* __restrict__ outp) {
  long n = (long)BATCH * GNY * GNX * 64;
  long i = (long)blockIdx.x * blockDim.x + threadIdx.x;
  long stride = (long)gridDim.x * blockDim.x;
  for (; i < n; i += stride) {
    int ch = (int)(i & 63);
    int x = (int)((i >> 6) & 255);
    int y = (int)((i >> 14) & 255);
    int b = (int)(i >> 22);
    float v = fmaxf(fmaf(raw[i], scsh[ch * 2], scsh[ch * 2 + 1]), 0.f);
    outp[(((size_t)b * 64 + ch) * GNY + y) * GNX + x] = v;   // NCHW
  }
}

// ------------------------------- launcher ------------------------------------

static void fill_u32(unsigned* p, unsigned v, long n, hipStream_t s) {
  long blocks = (n + 255) / 256;
  if (blocks > 16384) blocks = 16384;
  k_fill_u32<<<dim3((unsigned)blocks), dim3(256), 0, s>>>(p, v, n);
}

extern "C" void kernel_launch(void* const* d_in, const int* in_sizes, int n_in,
                              void* d_out, int out_size, void* d_ws, size_t ws_size,
                              hipStream_t stream) {
  (void)in_sizes; (void)n_in; (void)out_size; (void)ws_size;
  const float* pts = (const float*)d_in[0];
  const unsigned char* msk = (const unsigned char*)d_in[1];
  const float* w1 = (const float*)d_in[2];
  const float* b1 = (const float*)d_in[3];
  const float* g1 = (const float*)d_in[4];
  const float* be1 = (const float*)d_in[5];
  const float* w2 = (const float*)d_in[6];
  const float* b2 = (const float*)d_in[7];
  const float* g2 = (const float*)d_in[8];
  const float* be2 = (const float*)d_in[9];

  // workspace carve-out (~530 MB)
  char* w = (char*)d_ws;
  auto carve = [&](size_t bytes) -> void* {
    void* p = (void*)w;
    w += (bytes + 255) & ~(size_t)255;
    return p;
  };
  int*      pid    = (int*)carve(sizeof(int) * BATCH * NPTS);
  int*      pslot  = (int*)carve(sizeof(int) * BATCH * NPTS);
  int*      prank  = (int*)carve(sizeof(int) * BATCH * NPTS);
  unsigned* occ    = (unsigned*)carve(sizeof(unsigned) * BATCH * SENTID);
  int*      c2s    = (int*)carve(sizeof(int) * BATCH * SENTID);
  int*      s2c    = (int*)carve(sizeof(int) * BATCH * MAXP);
  int*      arrive = (int*)carve(sizeof(int) * BATCH * MAXP);
  int*      kcnt   = (int*)carve(sizeof(int) * BATCH * MAXP);
  float*    sums   = (float*)carve(sizeof(float) * BATCH * MAXP * 3);
  unsigned* pm     = (unsigned*)carve(sizeof(unsigned) * BATCH * MAXP);
  float*    pf     = (float*)carve(sizeof(float) * (size_t)BATCH * MAXP * MAXPP * 10);
  float*    y1     = (float*)carve(sizeof(float) * (size_t)BATCH * ROWS * 64);
  float*    y2     = (float*)carve(sizeof(float) * (size_t)BATCH * ROWS * 64);
  _Float16* w1h    = (_Float16*)carve(sizeof(_Float16) * 64 * 32);
  _Float16* w2h    = (_Float16*)carve(sizeof(_Float16) * 64 * 64);
  _Float16* cwh    = (_Float16*)carve(sizeof(_Float16) * 3 * 9 * 64 * 64);
  float*    bnreg  = (float*)carve(sizeof(float) * 4096);
  float*    actA   = (float*)carve(sizeof(float) * (size_t)BATCH * PADW * PADW * 64);
  float*    actB   = (float*)carve(sizeof(float) * (size_t)BATCH * PADW * PADW * 64);
  float*    raw    = (float*)carve(sizeof(float) * (size_t)BATCH * GNY * GNX * 64);

  float* bn1acc = bnreg + 0;     // B*64*2
  float* bn1p   = bnreg + 256;
  float* bn2acc = bnreg + 512;
  float* bn2p   = bnreg + 768;

  // per-call zero init (read-before-write regions only)
  fill_u32(occ, 0u, (long)BATCH * SENTID, stream);
  fill_u32((unsigned*)s2c, 0xFFFFFFFFu, (long)BATCH * MAXP, stream);
  fill_u32((unsigned*)arrive, 0u, (long)BATCH * MAXP, stream);
  fill_u32((unsigned*)kcnt, 0u, (long)BATCH * MAXP, stream);
  fill_u32((unsigned*)sums, 0u, (long)BATCH * MAXP * 3, stream);
  fill_u32(pm, 0u, (long)BATCH * MAXP, stream);
  fill_u32((unsigned*)pf, 0u, (long)BATCH * MAXP * MAXPP * 10, stream);
  fill_u32((unsigned*)bnreg, 0u, 4096, stream);
  fill_u32((unsigned*)actA, 0u, (long)BATCH * PADW * PADW * 64, stream);
  fill_u32((unsigned*)actB, 0u, (long)BATCH * PADW * PADW * 64, stream);

  const int nptsBlocks = (BATCH * NPTS + 255) / 256;
  k1_assign<<<nptsBlocks, 256, 0, stream>>>(pts, msk, pid, occ);
  k2_scan<<<BATCH, 1024, 0, stream>>>(occ, c2s, s2c);
  k3_rank<<<nptsBlocks, 256, 0, stream>>>(pts, pid, c2s, arrive, kcnt, sums, pslot, prank);
  k4_features<<<nptsBlocks, 256, 0, stream>>>(pts, pid, pslot, prank, kcnt, sums, pf, pm);

  kw_prep<<<(2048 + 4096 + 3 * 9 * 4096 + 255) / 256, 256, 0, stream>>>(
      w1, w2, (const float*)d_in[10], (const float*)d_in[14], (const float*)d_in[18],
      w1h, w2h, cwh);

  // pillar MLP (per-sample BN stats)
  k5_gemm1<<<dim3(TILES16 / 4, BATCH), 128, 0, stream>>>(pf, w1h, b1, y1, bn1acc);
  k_bnfin<<<1, 128, 0, stream>>>(bn1acc, g1, be1, bn1p, 1.0f / (float)ROWS, BATCH * 64);
  k7_gemm2<<<dim3(TILES16 / 4, BATCH), 128, 0, stream>>>(y1, w2h, b2, bn1p, y2, bn2acc);
  k_bnfin<<<1, 128, 0, stream>>>(bn2acc, g2, be2, bn2p, 1.0f / (float)ROWS, BATCH * 64);
  k9_maxscatter<<<dim3(MAXP, BATCH), 64, 0, stream>>>(y2, pm, s2c, bn2p, actA);

  // conv stack: actA -> actB -> actA -> d_out
  const float* convIn[3] = {actA, actB, actA};
  float* convPost[2] = {actB, actA};
  for (int l = 0; l < 3; ++l) {
    const _Float16* wl = cwh + (size_t)l * 9 * 64 * 64;
    const float* cb = (const float*)d_in[11 + 4 * l];
    const float* cg = (const float*)d_in[12 + 4 * l];
    const float* cbe = (const float*)d_in[13 + 4 * l];
    float* acc = bnreg + 1024 + l * 128;
    float* par = bnreg + 1408 + l * 128;
    k10_conv<<<dim3(GNX / 16, GNY / 8, BATCH), 256, 0, stream>>>(convIn[l], wl, cb, raw, acc);
    k_bnfin<<<1, 64, 0, stream>>>(acc, cg, cbe, par, 1.0f / (float)(BATCH * GNY * GNX), 64);
    if (l < 2)
      k11_apply_pad<<<4096, 256, 0, stream>>>(raw, par, convPost[l]);
    else
      k11_apply_out<<<4096, 256, 0, stream>>>(raw, par, (float*)d_out);
  }
}
